// ProteinEncoder_62672162783751
// MI455X (gfx1250) — matrix operand order repaired and checked by reading, not compiled, not run on hardware
//
#include <hip/hip_runtime.h>

// ---------------------------------------------------------------------------
// EGNN forward for MI455X (gfx1250, wave32).
// Edge/node MLPs run on v_wmma_f32_16x16x32_f16 (f16 operands, f32 accum).
// LDS stages transposed f16 weights; a per-wave LDS tile performs the
// C-layout -> A-layout transpose between the two GEMM layers, ordered with
// s_wait_dscnt. Scatter-sums use global_atomic_add_f32.
// ---------------------------------------------------------------------------

#define NN 20000
#define NE 640000
#define WPB 8            // waves per 256-thread block

typedef __attribute__((ext_vector_type(16))) _Float16 v16h;
typedef __attribute__((ext_vector_type(8)))  float    v8f;

union AF { v16h v; _Float16 h[16]; uint4 u[2]; };
union CF { v8f  v; float f[8]; };

__device__ __forceinline__ float silu_f(float x)    { return x / (1.0f + __expf(-x)); }
__device__ __forceinline__ float sigmoid_f(float x) { return 1.0f / (1.0f + __expf(-x)); }

// ---------------- small elementwise kernels ----------------

__global__ void k_tr(const float* __restrict__ src, _Float16* __restrict__ dst, int K) {
  int i = blockIdx.x * blockDim.x + threadIdx.x;        // i = n*K + k
  if (i >= K * 64) return;
  int n = i / K, k = i - n * K;
  dst[i] = (_Float16)src[k * 64 + n];                   // W[k][n] -> WT[n][k]
}

__global__ void k_embed(const float* __restrict__ atom, const float* __restrict__ w,
                        const float* __restrict__ b, float* __restrict__ h,
                        _Float16* __restrict__ h16) {
  int i = blockIdx.x * blockDim.x + threadIdx.x;
  if (i >= NN * 64) return;
  int n = i >> 6, o = i & 63;
  const float* ar = atom + n * 31;
  float s = b[o];
#pragma unroll
  for (int k = 0; k < 31; k++) s = fmaf(ar[k], w[k * 64 + o], s);
  h[i] = s; h16[i] = (_Float16)s;
}

__global__ void k_out(const float* __restrict__ h, const float* __restrict__ w,
                      const float* __restrict__ b, float* __restrict__ out) {
  int i = blockIdx.x * blockDim.x + threadIdx.x;
  if (i >= NN * 64) return;
  int n = i >> 6, o = i & 63;
  const float* hr = h + (size_t)n * 64;
  float s = b[o];
#pragma unroll 8
  for (int k = 0; k < 64; k++) s = fmaf(hr[k], w[k * 64 + o], s);
  out[i] = s;
}

__global__ void k_dist0(const float* __restrict__ pos, const int* __restrict__ rows,
                        const int* __restrict__ cols, float* __restrict__ dist0) {
  int e = blockIdx.x * blockDim.x + threadIdx.x;
  if (e >= NE) return;
  int r = rows[e], c = cols[e];
  float dx = pos[r*3+0] - pos[c*3+0];
  float dy = pos[r*3+1] - pos[c*3+1];
  float dz = pos[r*3+2] - pos[c*3+2];
  dist0[e] = dx*dx + dy*dy + dz*dz;
}

__global__ void k_geom(const float* __restrict__ x, const int* __restrict__ rows,
                       const int* __restrict__ cols, const float* __restrict__ dist0,
                       float2* __restrict__ ea2) {
  int e = blockIdx.x * blockDim.x + threadIdx.x;
  if (e >= NE) return;
  int r = rows[e], c = cols[e];
  float dx = x[r*3+0] - x[c*3+0];
  float dy = x[r*3+1] - x[c*3+1];
  float dz = x[r*3+2] - x[c*3+2];
  ea2[e] = make_float2(dx*dx + dy*dy + dz*dz, dist0[e]);
}

__global__ void k_copy(const float* __restrict__ s, float* __restrict__ d, int n) {
  int i = blockIdx.x * blockDim.x + threadIdx.x;
  if (i < n) d[i] = s[i];
}

__global__ void k_axpy(float* __restrict__ x, const float* __restrict__ a, int n) {
  int i = blockIdx.x * blockDim.x + threadIdx.x;
  if (i < n) x[i] += a[i];
}

// ---------------- WMMA helpers ----------------

__device__ __forceinline__ void loadA_f32(AF& a, const float* p) {
  float4 x0 = *(const float4*)(p);
  float4 x1 = *(const float4*)(p + 4);
  float4 x2 = *(const float4*)(p + 16);
  float4 x3 = *(const float4*)(p + 20);
  a.h[0]=(_Float16)x0.x;  a.h[1]=(_Float16)x0.y;  a.h[2]=(_Float16)x0.z;  a.h[3]=(_Float16)x0.w;
  a.h[4]=(_Float16)x1.x;  a.h[5]=(_Float16)x1.y;  a.h[6]=(_Float16)x1.z;  a.h[7]=(_Float16)x1.w;
  a.h[8]=(_Float16)x2.x;  a.h[9]=(_Float16)x2.y;  a.h[10]=(_Float16)x2.z; a.h[11]=(_Float16)x2.w;
  a.h[12]=(_Float16)x3.x; a.h[13]=(_Float16)x3.y; a.h[14]=(_Float16)x3.z; a.h[15]=(_Float16)x3.w;
}

// ---------------- GCL edge kernel ----------------
// Per 16-edge tile: mij = silu(silu([h_r|h_c|ea] @ W1 + b1) @ W2 + b2),
// att = sigmoid(mij @ w_att + b_att), agg[row] += mij*att (atomics).

__global__ __launch_bounds__(256)
void k_gcl_edge(const _Float16* __restrict__ h16,
                const int* __restrict__ rows, const int* __restrict__ cols,
                const float2* __restrict__ ea2,
                const _Float16* __restrict__ w1t, const float* __restrict__ b1,
                const float* __restrict__ wea,      // e1.w rows 128..129 (f32)
                const _Float16* __restrict__ w2t, const float* __restrict__ b2,
                const float* __restrict__ attw, const float* __restrict__ attb,
                float* __restrict__ agg, int ntiles) {
  __shared__ _Float16 sW1[128 * 64];
  __shared__ _Float16 sW2[64 * 64];
  __shared__ _Float16 sT[WPB][16 * 64];

  for (int i = threadIdx.x; i < 128 * 64 / 8; i += blockDim.x)
    ((uint4*)sW1)[i] = ((const uint4*)w1t)[i];
  for (int i = threadIdx.x; i < 64 * 64 / 8; i += blockDim.x)
    ((uint4*)sW2)[i] = ((const uint4*)w2t)[i];
  __syncthreads();

  const int lane = threadIdx.x & 31;
  const int nmod = lane & 15;
  const int hi   = lane >> 4;
  const int ko   = hi ? 8 : 0;
  _Float16* myT = sT[threadIdx.x >> 5];
  const int gw = blockIdx.x * WPB + (threadIdx.x >> 5);
  const int nw = gridDim.x * WPB;
  const float ab = attb[0];

  for (int tile = gw; tile < ntiles; tile += nw) {
    const int e0 = tile * 16;
    const int eA = e0 + nmod;                       // A-matrix row owned by lane
    const _Float16* hr = h16 + (size_t)rows[eA] * 64;
    const _Float16* hc = h16 + (size_t)cols[eA] * 64;
    AF a[4];
#pragma unroll
    for (int c = 0; c < 2; c++) {
      a[c].u[0]     = *(const uint4*)(hr + c * 32 + ko);
      a[c].u[1]     = *(const uint4*)(hr + c * 32 + ko + 16);
      a[2 + c].u[0] = *(const uint4*)(hc + c * 32 + ko);
      a[2 + c].u[1] = *(const uint4*)(hc + c * 32 + ko + 16);
    }
    float dloc[8], d0loc[8]; int rloc[8];
#pragma unroll
    for (int j = 0; j < 8; j++) {
      int e = e0 + hi * 8 + j;
      float2 t = ea2[e];
      dloc[j] = t.x; d0loc[j] = t.y; rloc[j] = rows[e];
    }
    // layer 1: 4 N-tiles x 4 K-chunks, edge-attr folded into the bias
#pragma unroll
    for (int t = 0; t < 4; t++) {
      const int n = t * 16 + nmod;
      const float bb = b1[n], wd = wea[n], w0 = wea[64 + n];
      CF acc;
#pragma unroll
      for (int j = 0; j < 8; j++) acc.f[j] = fmaf(dloc[j], wd, fmaf(d0loc[j], w0, bb));
#pragma unroll
      for (int c = 0; c < 4; c++) {
        AF bf;
        const _Float16* bp = sW1 + n * 128 + c * 32 + (hi ? 8 : 0);
        bf.u[0] = *(const uint4*)bp; bf.u[1] = *(const uint4*)(bp + 16);
        acc.v = __builtin_amdgcn_wmma_f32_16x16x32_f16(false, a[c].v, false, bf.v,
                                                       (short)0, acc.v, false, false);
      }
#pragma unroll
      for (int j = 0; j < 8; j++)
        myT[(hi * 8 + j) * 64 + n] = (_Float16)silu_f(acc.f[j]);
    }
    asm volatile("s_wait_dscnt 0x0" ::: "memory");   // wave-private LDS transpose
    AF a2[2];
    const _Float16* tp = myT + nmod * 64;
#pragma unroll
    for (int c = 0; c < 2; c++) {
      a2[c].u[0] = *(const uint4*)(tp + c * 32 + ko);
      a2[c].u[1] = *(const uint4*)(tp + c * 32 + ko + 16);
    }
    // layer 2 + attention logits
    CF m2[4]; float sj[8];
#pragma unroll
    for (int j = 0; j < 8; j++) sj[j] = 0.f;
#pragma unroll
    for (int t = 0; t < 4; t++) {
      const int n = t * 16 + nmod;
      CF acc;
#pragma unroll
      for (int j = 0; j < 8; j++) acc.f[j] = b2[n];
#pragma unroll
      for (int c = 0; c < 2; c++) {
        AF bf;
        const _Float16* bp = sW2 + n * 64 + c * 32 + (hi ? 8 : 0);
        bf.u[0] = *(const uint4*)bp; bf.u[1] = *(const uint4*)(bp + 16);
        acc.v = __builtin_amdgcn_wmma_f32_16x16x32_f16(false, a2[c].v, false, bf.v,
                                                       (short)0, acc.v, false, false);
      }
      const float wn = attw[n];
#pragma unroll
      for (int j = 0; j < 8; j++) {
        float s = silu_f(acc.f[j]);
        m2[t].f[j] = s;
        sj[j] = fmaf(s, wn, sj[j]);
      }
    }
#pragma unroll
    for (int off = 1; off < 16; off <<= 1) {
#pragma unroll
      for (int j = 0; j < 8; j++) sj[j] += __shfl_xor(sj[j], off, 32);
    }
    float aj[8];
#pragma unroll
    for (int j = 0; j < 8; j++) aj[j] = sigmoid_f(sj[j] + ab);
    // scatter-sum into agg[row]
#pragma unroll
    for (int t = 0; t < 4; t++) {
      const int n = t * 16 + nmod;
#pragma unroll
      for (int j = 0; j < 8; j++)
        atomicAdd(agg + (size_t)rloc[j] * 64 + n, m2[t].f[j] * aj[j]);
    }
  }
}

// ---------------- GCL node kernel ----------------
// h = h + silu([h|agg] @ n1 + b1) @ n2 + b2 ; refresh h16.

__global__ __launch_bounds__(256)
void k_gcl_node(float* __restrict__ h, const float* __restrict__ agg,
                const _Float16* __restrict__ w1t, const float* __restrict__ b1,
                const _Float16* __restrict__ w2t, const float* __restrict__ b2,
                _Float16* __restrict__ h16, int ntiles) {
  __shared__ _Float16 sW1[128 * 64];
  __shared__ _Float16 sW2[64 * 64];
  __shared__ _Float16 sT[WPB][16 * 64];

  for (int i = threadIdx.x; i < 128 * 64 / 8; i += blockDim.x)
    ((uint4*)sW1)[i] = ((const uint4*)w1t)[i];
  for (int i = threadIdx.x; i < 64 * 64 / 8; i += blockDim.x)
    ((uint4*)sW2)[i] = ((const uint4*)w2t)[i];
  __syncthreads();

  const int lane = threadIdx.x & 31;
  const int nmod = lane & 15;
  const int hi   = lane >> 4;
  const int ko   = hi ? 8 : 0;
  _Float16* myT = sT[threadIdx.x >> 5];
  const int gw = blockIdx.x * WPB + (threadIdx.x >> 5);
  const int nw = gridDim.x * WPB;

  for (int tile = gw; tile < ntiles; tile += nw) {
    const int m = tile * 16 + nmod;
    const float* hp = h   + (size_t)m * 64;
    const float* gp = agg + (size_t)m * 64;
    AF a[4];
    loadA_f32(a[0], hp + ko);      loadA_f32(a[1], hp + 32 + ko);
    loadA_f32(a[2], gp + ko);      loadA_f32(a[3], gp + 32 + ko);
#pragma unroll
    for (int t = 0; t < 4; t++) {
      const int n = t * 16 + nmod;
      CF acc;
#pragma unroll
      for (int j = 0; j < 8; j++) acc.f[j] = b1[n];
#pragma unroll
      for (int c = 0; c < 4; c++) {
        AF bf;
        const _Float16* bp = sW1 + n * 128 + c * 32 + (hi ? 8 : 0);
        bf.u[0] = *(const uint4*)bp; bf.u[1] = *(const uint4*)(bp + 16);
        acc.v = __builtin_amdgcn_wmma_f32_16x16x32_f16(false, a[c].v, false, bf.v,
                                                       (short)0, acc.v, false, false);
      }
#pragma unroll
      for (int j = 0; j < 8; j++)
        myT[(hi * 8 + j) * 64 + n] = (_Float16)silu_f(acc.f[j]);
    }
    asm volatile("s_wait_dscnt 0x0" ::: "memory");
    AF a2[2];
    const _Float16* tp = myT + nmod * 64;
#pragma unroll
    for (int c = 0; c < 2; c++) {
      a2[c].u[0] = *(const uint4*)(tp + c * 32 + ko);
      a2[c].u[1] = *(const uint4*)(tp + c * 32 + ko + 16);
    }
#pragma unroll
    for (int t = 0; t < 4; t++) {
      const int n = t * 16 + nmod;
      CF acc;
#pragma unroll
      for (int j = 0; j < 8; j++) acc.f[j] = b2[n];
#pragma unroll
      for (int c = 0; c < 2; c++) {
        AF bf;
        const _Float16* bp = sW2 + n * 64 + c * 32 + (hi ? 8 : 0);
        bf.u[0] = *(const uint4*)bp; bf.u[1] = *(const uint4*)(bp + 16);
        acc.v = __builtin_amdgcn_wmma_f32_16x16x32_f16(false, a2[c].v, false, bf.v,
                                                       (short)0, acc.v, false, false);
      }
#pragma unroll
      for (int j = 0; j < 8; j++) {
        size_t idx = (size_t)(tile * 16 + hi * 8 + j) * 64 + n;
        float v = h[idx] + acc.f[j];                 // residual, no outer act
        h[idx] = v; h16[idx] = (_Float16)v;
      }
    }
  }
}

// ---------------- coord-update kernel ----------------
// phi = silu(silu([h_r|h_c|ea] @ c1 + b1) @ c2 + b2) @ c3 ; xacc[row] += coord_diff*phi

__global__ __launch_bounds__(256)
void k_coord(const _Float16* __restrict__ h16,
             const int* __restrict__ rows, const int* __restrict__ cols,
             const float2* __restrict__ ea2,
             const _Float16* __restrict__ w1t, const float* __restrict__ b1,
             const float* __restrict__ wea,
             const _Float16* __restrict__ w2t, const float* __restrict__ b2,
             const float* __restrict__ c3w,
             const float* __restrict__ x, float* __restrict__ xacc, int ntiles) {
  __shared__ _Float16 sW1[128 * 64];
  __shared__ _Float16 sW2[64 * 64];
  __shared__ _Float16 sT[WPB][16 * 64];

  for (int i = threadIdx.x; i < 128 * 64 / 8; i += blockDim.x)
    ((uint4*)sW1)[i] = ((const uint4*)w1t)[i];
  for (int i = threadIdx.x; i < 64 * 64 / 8; i += blockDim.x)
    ((uint4*)sW2)[i] = ((const uint4*)w2t)[i];
  __syncthreads();

  const int lane = threadIdx.x & 31;
  const int nmod = lane & 15;
  const int hi   = lane >> 4;
  const int ko   = hi ? 8 : 0;
  _Float16* myT = sT[threadIdx.x >> 5];
  const int gw = blockIdx.x * WPB + (threadIdx.x >> 5);
  const int nw = gridDim.x * WPB;

  for (int tile = gw; tile < ntiles; tile += nw) {
    const int e0 = tile * 16;
    const int eA = e0 + nmod;
    const _Float16* hr = h16 + (size_t)rows[eA] * 64;
    const _Float16* hc = h16 + (size_t)cols[eA] * 64;
    AF a[4];
#pragma unroll
    for (int c = 0; c < 2; c++) {
      a[c].u[0]     = *(const uint4*)(hr + c * 32 + ko);
      a[c].u[1]     = *(const uint4*)(hr + c * 32 + ko + 16);
      a[2 + c].u[0] = *(const uint4*)(hc + c * 32 + ko);
      a[2 + c].u[1] = *(const uint4*)(hc + c * 32 + ko + 16);
    }
    float dloc[8], d0loc[8]; int rloc[8];
#pragma unroll
    for (int j = 0; j < 8; j++) {
      int e = e0 + hi * 8 + j;
      float2 t = ea2[e];
      dloc[j] = t.x; d0loc[j] = t.y; rloc[j] = rows[e];
    }
#pragma unroll
    for (int t = 0; t < 4; t++) {
      const int n = t * 16 + nmod;
      const float bb = b1[n], wd = wea[n], w0 = wea[64 + n];
      CF acc;
#pragma unroll
      for (int j = 0; j < 8; j++) acc.f[j] = fmaf(dloc[j], wd, fmaf(d0loc[j], w0, bb));
#pragma unroll
      for (int c = 0; c < 4; c++) {
        AF bf;
        const _Float16* bp = sW1 + n * 128 + c * 32 + (hi ? 8 : 0);
        bf.u[0] = *(const uint4*)bp; bf.u[1] = *(const uint4*)(bp + 16);
        acc.v = __builtin_amdgcn_wmma_f32_16x16x32_f16(false, a[c].v, false, bf.v,
                                                       (short)0, acc.v, false, false);
      }
#pragma unroll
      for (int j = 0; j < 8; j++)
        myT[(hi * 8 + j) * 64 + n] = (_Float16)silu_f(acc.f[j]);
    }
    asm volatile("s_wait_dscnt 0x0" ::: "memory");
    AF a2[2];
    const _Float16* tp = myT + nmod * 64;
#pragma unroll
    for (int c = 0; c < 2; c++) {
      a2[c].u[0] = *(const uint4*)(tp + c * 32 + ko);
      a2[c].u[1] = *(const uint4*)(tp + c * 32 + ko + 16);
    }
    float sj[8];
#pragma unroll
    for (int j = 0; j < 8; j++) sj[j] = 0.f;
#pragma unroll
    for (int t = 0; t < 4; t++) {
      const int n = t * 16 + nmod;
      CF acc;
#pragma unroll
      for (int j = 0; j < 8; j++) acc.f[j] = b2[n];
#pragma unroll
      for (int c = 0; c < 2; c++) {
        AF bf;
        const _Float16* bp = sW2 + n * 64 + c * 32 + (hi ? 8 : 0);
        bf.u[0] = *(const uint4*)bp; bf.u[1] = *(const uint4*)(bp + 16);
        acc.v = __builtin_amdgcn_wmma_f32_16x16x32_f16(false, a2[c].v, false, bf.v,
                                                       (short)0, acc.v, false, false);
      }
      const float wn = c3w[n];
#pragma unroll
      for (int j = 0; j < 8; j++) sj[j] = fmaf(silu_f(acc.f[j]), wn, sj[j]);
    }
#pragma unroll
    for (int off = 1; off < 16; off <<= 1) {
#pragma unroll
      for (int j = 0; j < 8; j++) sj[j] += __shfl_xor(sj[j], off, 32);
    }
    if (nmod < 8) {                  // 16 active lanes <-> 16 edges
      int j = nmod;
      int e = e0 + hi * 8 + j;
      int r = rloc[j], c = cols[e];
      float dx = x[r*3+0] - x[c*3+0];
      float dy = x[r*3+1] - x[c*3+1];
      float dz = x[r*3+2] - x[c*3+2];
      float scl = sj[j] / (sqrtf(dloc[j] + 1e-8f) + 1.0f);   // norm_constant = 1
      atomicAdd(xacc + r*3+0, dx * scl);
      atomicAdd(xacc + r*3+1, dy * scl);
      atomicAdd(xacc + r*3+2, dz * scl);
    }
  }
}

// ---------------- host side ----------------

struct GclW { const float *attb,*attw,*e1b,*e1w,*e2b,*e2w,*n1b,*n1w,*n2b,*n2w; };
struct BlkW { const float *c1b,*c1w,*c2b,*c2w,*c3w; GclW g[3]; };

extern "C" void kernel_launch(void* const* d_in, const int* in_sizes, int n_in,
                              void* d_out, int out_size, void* d_ws, size_t ws_size,
                              hipStream_t stream) {
  (void)in_sizes; (void)n_in; (void)out_size; (void)ws_size;
  const float* atom = (const float*)d_in[0];
  const float* pos  = (const float*)d_in[1];
  const int*   rows = (const int*)d_in[2];
  const int*   cols = rows + NE;

  // params: jax pytree leaf order (dict keys sorted)
  BlkW blk[2]; int i = 3;
  for (int b = 0; b < 2; b++) {
    blk[b].c1b = (const float*)d_in[i++]; blk[b].c1w = (const float*)d_in[i++];
    blk[b].c2b = (const float*)d_in[i++]; blk[b].c2w = (const float*)d_in[i++];
    blk[b].c3w = (const float*)d_in[i++];
    for (int g = 0; g < 3; g++) {
      GclW& G = blk[b].g[g];
      G.attb = (const float*)d_in[i++]; G.attw = (const float*)d_in[i++];
      G.e1b  = (const float*)d_in[i++]; G.e1w  = (const float*)d_in[i++];
      G.e2b  = (const float*)d_in[i++]; G.e2w  = (const float*)d_in[i++];
      G.n1b  = (const float*)d_in[i++]; G.n1w  = (const float*)d_in[i++];
      G.n2b  = (const float*)d_in[i++]; G.n2w  = (const float*)d_in[i++];
    }
  }
  const float* embB = (const float*)d_in[i++];
  const float* embW = (const float*)d_in[i++];
  const float* outB = (const float*)d_in[i++];
  const float* outW = (const float*)d_in[i++];

  // workspace carve (256B aligned)
  char* p = (char*)d_ws;
  auto take = [&](size_t bytes) -> void* {
    void* r = (void*)p; p += (bytes + 255) & ~(size_t)255; return r;
  };
  float*    h     = (float*)take((size_t)NN * 64 * 4);
  float*    agg   = (float*)take((size_t)NN * 64 * 4);
  float2*   ea2   = (float2*)take((size_t)NE * 8);
  float*    dist0 = (float*)take((size_t)NE * 4);
  float*    x     = (float*)take((size_t)NN * 3 * 4);
  float*    xacc  = (float*)take((size_t)NN * 3 * 4);
  _Float16* h16   = (_Float16*)take((size_t)NN * 64 * 2);
  _Float16* twp   = (_Float16*)take((size_t)(6*(8192+4096+8192+4096) + 2*(8192+4096)) * 2);

  _Float16 *te1[6], *te2[6], *tn1[6], *tn2[6], *tc1[2], *tc2[2];
  {
    _Float16* q = twp;
    for (int b = 0; b < 2; b++) {
      tc1[b] = q; q += 8192; tc2[b] = q; q += 4096;
      for (int g = 0; g < 3; g++) {
        int k = b * 3 + g;
        te1[k] = q; q += 8192; te2[k] = q; q += 4096;
        tn1[k] = q; q += 8192; tn2[k] = q; q += 4096;
      }
    }
  }

  // f16 transposed weights (first 128 rows of the 130-row e1/c1)
  auto tr = [&](const float* src, _Float16* dst, int K) {
    k_tr<<<(K * 64 + 255) / 256, 256, 0, stream>>>(src, dst, K);
  };
  for (int b = 0; b < 2; b++) {
    tr(blk[b].c1w, tc1[b], 128); tr(blk[b].c2w, tc2[b], 64);
    for (int g = 0; g < 3; g++) {
      int k = b * 3 + g; GclW& G = blk[b].g[g];
      tr(G.e1w, te1[k], 128); tr(G.e2w, te2[k], 64);
      tr(G.n1w, tn1[k], 128); tr(G.n2w, tn2[k], 64);
    }
  }

  k_embed<<<(NN * 64 + 255) / 256, 256, 0, stream>>>(atom, embW, embB, h, h16);
  k_dist0<<<(NE + 255) / 256, 256, 0, stream>>>(pos, rows, cols, dist0);
  k_copy<<<(NN * 3 + 255) / 256, 256, 0, stream>>>(pos, x, NN * 3);

  const int etiles = NE / 16, ntiles = NN / 16;
  for (int b = 0; b < 2; b++) {
    k_geom<<<(NE + 255) / 256, 256, 0, stream>>>(x, rows, cols, dist0, ea2);
    for (int g = 0; g < 3; g++) {
      int k = b * 3 + g; GclW& G = blk[b].g[g];
      hipMemsetAsync(agg, 0, (size_t)NN * 64 * 4, stream);
      k_gcl_edge<<<512, 256, 0, stream>>>(h16, rows, cols, ea2,
                                          te1[k], G.e1b, G.e1w + 128 * 64,
                                          te2[k], G.e2b, G.attw, G.attb,
                                          agg, etiles);
      k_gcl_node<<<160, 256, 0, stream>>>(h, agg, tn1[k], G.n1b,
                                          tn2[k], G.n2b, h16, ntiles);
    }
    hipMemsetAsync(xacc, 0, (size_t)NN * 3 * 4, stream);
    k_coord<<<512, 256, 0, stream>>>(h16, rows, cols, ea2,
                                     tc1[b], blk[b].c1b, blk[b].c1w + 128 * 64,
                                     tc2[b], blk[b].c2b, blk[b].c3w,
                                     x, xacc, etiles);
    k_axpy<<<(NN * 3 + 255) / 256, 256, 0, stream>>>(x, xacc, NN * 3);
  }

  float* out = (float*)d_out;
  k_out<<<(NN * 64 + 255) / 256, 256, 0, stream>>>(h, outW, outB, out);
  k_copy<<<(NN * 3 + 255) / 256, 256, 0, stream>>>(x, out + (size_t)NN * 64, NN * 3);
}
